// LinearLayerTT_4629974745536
// MI455X (gfx1250) — matrix operand order repaired and checked by reading, not compiled, not run on hardware
//
#include <hip/hip_runtime.h>

// CDNA5 / gfx1250, wave32. bf16 WMMA (16x16x32) with f32 accumulate.

typedef __attribute__((ext_vector_type(16))) __bf16 v16bf;
typedef __attribute__((ext_vector_type(8)))  float  v8f;

union ABu { uint4 q[2]; v16bf v; };

// Load a 32-byte WMMA operand as two 16-byte chunks (A: halves 0..7 = K kb..kb+7,
// halves 8..15 = K kb+8..kb+15; B: halves 0..15 = contiguous K kb..kb+15).
__device__ __forceinline__ v16bf ld_ab(const __bf16* p0, const __bf16* p1) {
  ABu t;
  t.q[0] = *(const uint4*)p0;
  t.q[1] = *(const uint4*)p1;
  return t.v;
}

#define WMMA_BF16(A, B, C) \
  __builtin_amdgcn_wmma_f32_16x16x32_bf16(false, (A), false, (B), (short)0, (C), false, false)

// -------- prep: transpose cores into bf16 A-operand row layouts (L2-resident) ------

__global__ void prep_core1(const float* __restrict__ c1, __bf16* __restrict__ c1t) {
  int o = blockIdx.x * blockDim.x + threadIdx.x;   // 0..262143
  int row = o >> 9;                                // n1*16 + s2
  int k   = o & 511;                               // s1*32 + m1
  int n1 = row >> 4, s2 = row & 15;
  int s1 = k >> 5,   m1 = k & 31;
  c1t[o] = (__bf16)c1[((s1 * 32 + n1) * 32 + m1) * 16 + s2];
}

__global__ void prep_core2(const float* __restrict__ c2, __bf16* __restrict__ c2t) {
  int o = blockIdx.x * blockDim.x + threadIdx.x;   // 0..16383
  int n2 = o >> 9;
  int k  = o & 511;                                // s2*32 + m2
  int s2 = k >> 5, m2 = k & 31;
  c2t[o] = (__bf16)c2[(s2 * 32 + n2) * 32 + m2];
}

// -------- fused TT kernel: one block (16 wave32s) per batch element --------

#define Z_PITCH 520   // 512 + 8 halfwords: 16B-aligned rows, bank-staggered

__global__ __launch_bounds__(512) void tt_fused(
    const float* __restrict__ x,                 // [256,32,32,32]
    const float* __restrict__ core0,             // [32(n0),32(m0),16(s1)]
    const __bf16* __restrict__ core1t,           // bf16 [512 rows (n1*16+s2)][512 (s1*32+m1)]
    const __bf16* __restrict__ core2t,           // bf16 [32 rows (n2)][512 (s2*32+m2)]
    const float* __restrict__ bias,              // [32,32,32]
    float* __restrict__ out)                     // [256,32,32,32]
{
  __shared__ __align__(16) __bf16 xt[1024 * 32];      // [p=(m1*32+m2)][m0]  64KB
  __shared__ __align__(16) __bf16 z1t[32 * Z_PITCH];  // [m2][s1*32+m1]      32.5KB
  __shared__ __align__(16) __bf16 z2t[32 * Z_PITCH];  // [n1][s2*32+m2]      32.5KB
  __shared__ __align__(16) __bf16 c0s[16 * 32];       // [s1][m0]            1KB
  __shared__ __align__(16) float  ys[32 * 32];        // [n1][n2]            4KB

  const int b    = blockIdx.x;
  const int tid  = threadIdx.x;
  const int lane = tid & 31;
  const int wave = tid >> 5;
  const int lrow = lane & 15;
  const int aoff = (lane < 16) ? 0 : 16;   // A-operand K base within 32-wide step
  const int boff = (lane < 16) ? 0 : 16;   // B-operand K base within 32-wide step
  const int ehi  = (lane < 16) ? 0 : 8;    // C/D row offset

  // ---- stage x_b into LDS as bf16, transposed to [p][m0] ----
  const float* xb = x + (size_t)b * 32768;
  for (int m0 = 0; m0 < 32; ++m0) {
#pragma unroll
    for (int rep = 0; rep < 2; ++rep) {
      int p = tid + rep * 512;
      xt[p * 32 + m0] = (__bf16)xb[m0 * 1024 + p];
    }
  }

  for (int n0 = 0; n0 < 32; ++n0) {
    __syncthreads();

    // c0 slice for this n0: c0s[s1][m0] = core0[n0][m0][s1]
    {
      int s1 = tid >> 5;         // 0..15
      int m0 = tid & 31;
      c0s[s1 * 32 + m0] = (__bf16)core0[n0 * 512 + m0 * 16 + s1];
    }
    __syncthreads();

    // ---- stage 1: z1(16x1024) = c0s(16x32) * xt(32x1024); 4 N-tiles per wave ----
    {
      v16bf a = ld_ab(&c0s[lrow * 32 + aoff], &c0s[lrow * 32 + aoff + 8]);
#pragma unroll
      for (int t = 0; t < 4; ++t) {
        int nt = wave * 4 + t;
        int p  = nt * 16 + lrow;                       // column = m1*32+m2
        v16bf bm = ld_ab(&xt[p * 32 + boff], &xt[p * 32 + boff + 8]);
        v8f c = {};
        c = WMMA_BF16(a, bm, c);
        int m1 = p >> 5, m2 = p & 31;
#pragma unroll
        for (int e = 0; e < 8; ++e) {
          int s1 = e + ehi;
          z1t[m2 * Z_PITCH + s1 * 32 + m1] = (__bf16)c[e];  // col-major for stage-2 B
        }
      }
    }
    __syncthreads();

    // ---- stage 2 (dominant): C(512x32) = core1t(512x512) * z1(512x32) ----
    // Each wave: 2 row-tiles x 2 col-tiles, K=512 -> 64 WMMAs.
    {
      v8f acc[2][2] = {};
      const int rt0 = wave * 2;
      for (int kk = 0; kk < 16; ++kk) {
        int k0 = kk * 32;
        const __bf16* a0p = &core1t[(size_t)(rt0 * 16 + lrow) * 512 + k0];
        const __bf16* a1p = &core1t[(size_t)((rt0 + 1) * 16 + lrow) * 512 + k0];
        v16bf a0 = ld_ab(a0p + aoff, a0p + aoff + 8);
        v16bf a1 = ld_ab(a1p + aoff, a1p + aoff + 8);
        v16bf b0 = ld_ab(&z1t[lrow * Z_PITCH + k0 + boff],
                         &z1t[lrow * Z_PITCH + k0 + boff + 8]);
        v16bf b1 = ld_ab(&z1t[(16 + lrow) * Z_PITCH + k0 + boff],
                         &z1t[(16 + lrow) * Z_PITCH + k0 + boff + 8]);
        acc[0][0] = WMMA_BF16(a0, b0, acc[0][0]);
        acc[0][1] = WMMA_BF16(a0, b1, acc[0][1]);
        acc[1][0] = WMMA_BF16(a1, b0, acc[1][0]);
        acc[1][1] = WMMA_BF16(a1, b1, acc[1][1]);
      }
#pragma unroll
      for (int i = 0; i < 2; ++i)
#pragma unroll
        for (int j = 0; j < 2; ++j)
#pragma unroll
          for (int e = 0; e < 8; ++e) {
            int grow = (rt0 + i) * 16 + e + ehi;       // n1*16 + s2
            int n1 = grow >> 4, s2 = grow & 15;
            int m2 = j * 16 + lrow;
            z2t[n1 * Z_PITCH + s2 * 32 + m2] = (__bf16)acc[i][j][e];  // col-major for stage-3 B
          }
    }
    __syncthreads();

    // ---- stage 3: Y(32x32) = core2t(32x512) * z2(512x32); waves 0..3 ----
    if (wave < 4) {
      const int mt  = wave >> 1;   // n2 tile
      const int nt2 = wave & 1;    // n1 tile
      v8f acc = {};
      for (int kk = 0; kk < 16; ++kk) {
        int k0 = kk * 32;
        const __bf16* ap = &core2t[(mt * 16 + lrow) * 512 + k0];
        v16bf a  = ld_ab(ap + aoff, ap + aoff + 8);
        v16bf bm = ld_ab(&z2t[(nt2 * 16 + lrow) * Z_PITCH + k0 + boff],
                         &z2t[(nt2 * 16 + lrow) * Z_PITCH + k0 + boff + 8]);
        acc = WMMA_BF16(a, bm, acc);
      }
#pragma unroll
      for (int e = 0; e < 8; ++e) {
        int n2 = mt * 16 + e + ehi;
        int n1 = nt2 * 16 + lrow;
        ys[n1 * 32 + n2] = acc[e];
      }
    }
    __syncthreads();

    // ---- coalesced epilogue: out[b,n0,:,:] = ys + bias[n0,:,:] ----
    if (tid < 128) {
      float* op       = out + (size_t)b * 32768 + n0 * 1024;
      const float* bp = bias + n0 * 1024;
#pragma unroll
      for (int i = 0; i < 8; ++i) {
        int o = tid * 8 + i;
        op[o] = ys[o] + bp[o];
      }
    }
  }
}

extern "C" void kernel_launch(void* const* d_in, const int* in_sizes, int n_in,
                              void* d_out, int out_size, void* d_ws, size_t ws_size,
                              hipStream_t stream) {
  (void)in_sizes; (void)n_in; (void)out_size; (void)ws_size;
  const float* x    = (const float*)d_in[0];
  const float* c0   = (const float*)d_in[1];
  const float* c1   = (const float*)d_in[2];
  const float* c2   = (const float*)d_in[3];
  const float* bias = (const float*)d_in[4];
  float* out = (float*)d_out;

  __bf16* c1t = (__bf16*)d_ws;          // 512*512 bf16 = 512KB
  __bf16* c2t = c1t + 512 * 512;        // 32*512 bf16 = 32KB

  prep_core1<<<512, 512, 0, stream>>>(c1, c1t);
  prep_core2<<<32, 512, 0, stream>>>(c2, c2t);
  tt_fused<<<256, 512, 0, stream>>>(x, c0, c1t, c2t, bias, out);
}
